// DelayedMLP_86431921865160
// MI455X (gfx1250) — compile-verified, matrix operand order checked by reading, not run on hardware
//
#include <hip/hip_runtime.h>
#include <hip/hip_bf16.h>

// Problem dims (match reference)
#define BB 32
#define SS 2048
#define II 256
#define HH 512
#define OO 256

typedef __attribute__((ext_vector_type(16))) __bf16 v16bf;
typedef __attribute__((ext_vector_type(8)))  float  v8f;

// ---------------------------------------------------------------------------
// WMMA helper: D = A(16x32 bf16) x B(32x16 bf16) + C(16x16 f32)
// ---------------------------------------------------------------------------
__device__ __forceinline__ v8f wmma_bf16(v16bf a, v16bf b, v8f c) {
    return __builtin_amdgcn_wmma_f32_16x16x32_bf16(
        /*neg_a=*/false, a, /*neg_b=*/false, b,
        /*c_mod=*/(short)0, c, /*reuse_a=*/false, /*reuse_b=*/false);
}

// Load a 16-lane x 16-element bf16 fragment (A rows = M, B rows = N against
// transposed [N][K] weights so K is contiguous per lane).
__device__ __forceinline__ v16bf load_frag(const __bf16* __restrict__ base,
                                           int ld, int row0, int k0) {
    const int l = threadIdx.x & 31;
    const int r = row0 + (l & 15);
    const int k = k0 + ((l >> 4) << 4);
    return *(const v16bf*)(base + (long)r * ld + k);
}

__device__ __forceinline__ float sigmoidf_(float v) {
    return 1.0f / (1.0f + __expf(-v));
}

// ---------------------------------------------------------------------------
// Kernel 0: convert + transpose weights f32 -> bf16 (one-time, tiny)
// ---------------------------------------------------------------------------
__global__ void convert_weights(const float* __restrict__ Wg,
                                const float* __restrict__ W1,
                                const float* __restrict__ W2,
                                const float* __restrict__ W3,
                                __bf16* __restrict__ Wgt,
                                __bf16* __restrict__ W1t,
                                __bf16* __restrict__ W2t,
                                __bf16* __restrict__ W3t) {
    const long i0 = (long)blockIdx.x * blockDim.x + threadIdx.x;
    const long stride = (long)gridDim.x * blockDim.x;
    for (long e = i0; e < (long)II * II; e += stride) {
        int k = (int)(e / II), n = (int)(e % II);
        Wgt[(long)n * II + k] = (__bf16)Wg[e];
    }
    for (long e = i0; e < (long)II * HH; e += stride) {
        int k = (int)(e / HH), n = (int)(e % HH);
        W1t[(long)n * II + k] = (__bf16)W1[e];
    }
    for (long e = i0; e < (long)HH * HH; e += stride) {
        int k = (int)(e / HH), n = (int)(e % HH);
        W2t[(long)n * HH + k] = (__bf16)W2[e];
    }
    for (long e = i0; e < (long)HH * OO; e += stride) {
        int k = (int)(e / OO), n = (int)(e % OO);
        W3t[(long)n * HH + k] = (__bf16)W3[e];
    }
}

// ---------------------------------------------------------------------------
// Kernel 1: delay-gate scan.
// 2 blocks x 512 threads (16 waves); block handles 16 batch rows. Wg resident
// in LDS. Double-buffered x staging: global loads for step t+1 are issued at
// the top of step t (hidden behind the WMMA chains), committed to the
// alternate LDS buffer before the end-of-step barrier, so the per-step
// critical path is pure barrier->WMMA->VALU. Two accumulators per matmul give
// 2-way WMMA ILP on the serialized K chain.
// Dynamic LDS: 128K Wg + 16K xt(x2) + 8K bufh + 16K buf_f32 = 168K
// ---------------------------------------------------------------------------
__global__ void __launch_bounds__(512, 1)
scan_kernel(const float* __restrict__ x,
            const float* __restrict__ bg,
            const __bf16* __restrict__ Wgt,
            __bf16* __restrict__ combined) {
    extern __shared__ __align__(32) char smem[];
    __bf16* sWg   = (__bf16*)smem;              // [256][256]
    __bf16* sXt   = sWg + II * II;              // 2 x [16][256]
    __bf16* sBufh = sXt + 2 * 16 * II;          // [16][256]
    float*  sBuf  = (float*)(sBufh + 16 * II);  // [16][256]

    const int tid  = threadIdx.x;
    const int wave = tid >> 5;
    const int lane = tid & 31;
    const int b0   = blockIdx.x * 16;

    // Stage Wg into LDS with 16B vectors (one-time, 128 KB).
    {
        const uint4* src = (const uint4*)Wgt;
        uint4* dst = (uint4*)sWg;
        const int nvec = (II * II * 2) / 16;
        for (int i = tid; i < nvec; i += blockDim.x) dst[i] = src[i];
    }
    for (int i = tid; i < 16 * II; i += blockDim.x) sBuf[i] = 0.0f;

    // Initial x tile (t = 0) into buffer 0.
    for (int i = tid; i < 16 * II; i += blockDim.x) {
        const int r = i >> 8, c = i & 255;
        sXt[i] = (__bf16)x[((long)(b0 + r) * SS + 0) * II + c];
    }

    const int n0 = wave * 16;           // this wave's N-tile
    const int nl = lane & 15;
    const int n  = n0 + nl;
    const float bias = bg[n];
    const int mhi = (lane >> 4) << 3;   // +8 for upper lane half

    __syncthreads();

    for (int t = 0; t < SS; ++t) {
        __bf16* sXtCur = sXt + (t & 1) * 16 * II;
        __bf16* sXtNxt = sXt + ((t + 1) & 1) * 16 * II;

        // Issue global loads for step t+1 now; they complete under the WMMAs.
        float xnext[8];
        const bool have_next = (t + 1 < SS);
        if (have_next) {
            #pragma unroll
            for (int q = 0; q < 8; ++q) {
                const int i = tid + q * 512;        // coalesced 512-wide chunks
                const int r = i >> 8, c = i & 255;
                xnext[q] = x[((long)(b0 + r) * SS + (t + 1)) * II + c];
            }
        }

        // ---- matmul 1: G1 tile = sigmoid(xt @ Wg + bg), 2 acc chains ----
        v8f acc0 = {0.f, 0.f, 0.f, 0.f, 0.f, 0.f, 0.f, 0.f};
        v8f acc1 = {0.f, 0.f, 0.f, 0.f, 0.f, 0.f, 0.f, 0.f};
        #pragma unroll
        for (int kk = 0; kk < II; kk += 64) {
            acc0 = wmma_bf16(load_frag(sXtCur, II, 0, kk),
                             load_frag(sWg, II, n0, kk), acc0);
            acc1 = wmma_bf16(load_frag(sXtCur, II, 0, kk + 32),
                             load_frag(sWg, II, n0, kk + 32), acc1);
        }

        float imm[8];
        #pragma unroll
        for (int e = 0; e < 8; ++e) {
            const int m = e + mhi;                 // C layout: VGPR e -> M
            const float d  = sigmoidf_(acc0[e] + acc1[e] + bias);
            const float xv = (float)sXtCur[m * II + n];
            const float bv = sBuf[m * II + n];
            imm[e] = xv * d;
            const float nb = fmaf(1.0f - d, xv, bv);
            sBuf[m * II + n]  = nb;
            sBufh[m * II + n] = (__bf16)nb;
        }
        __syncthreads();

        // ---- matmul 2: G2 tile = sigmoid(buf @ Wg + bg) ----
        v8f acc2 = {0.f, 0.f, 0.f, 0.f, 0.f, 0.f, 0.f, 0.f};
        v8f acc3 = {0.f, 0.f, 0.f, 0.f, 0.f, 0.f, 0.f, 0.f};
        #pragma unroll
        for (int kk = 0; kk < II; kk += 64) {
            acc2 = wmma_bf16(load_frag(sBufh, II, 0, kk),
                             load_frag(sWg, II, n0, kk), acc2);
            acc3 = wmma_bf16(load_frag(sBufh, II, 0, kk + 32),
                             load_frag(sWg, II, n0, kk + 32), acc3);
        }

        #pragma unroll
        for (int e = 0; e < 8; ++e) {
            const int m = e + mhi;
            const float bd = sigmoidf_(acc2[e] + acc3[e] + bias);
            const float bv = sBuf[m * II + n];
            sBuf[m * II + n] = bv * (1.0f - bd);
            const float outv = imm[e] + bv * bd;
            combined[((long)(b0 + m) * SS + t) * II + n] = (__bf16)outv;
        }

        // Commit next x tile to the alternate buffer (off the critical path:
        // only the LDS stores + barrier remain before step t+1's WMMAs).
        if (have_next) {
            #pragma unroll
            for (int q = 0; q < 8; ++q) {
                const int i = tid + q * 512;
                sXtNxt[i] = (__bf16)xnext[q];
            }
        }
        __syncthreads();
    }
}

// ---------------------------------------------------------------------------
// Kernel 2: fused 3-layer MLP over 64-token tiles.
// 1024 blocks x 256 threads (8 waves). Activations in LDS (160 KB); weights
// (1 MB bf16) stream from L2. Register-blocked over 4 M-tiles per N-tile:
// each B fragment is loaded once and reused by 4 independent WMMA chains.
// ---------------------------------------------------------------------------
__global__ void __launch_bounds__(256, 1)
mlp_kernel(const __bf16* __restrict__ combined,
           const __bf16* __restrict__ W1t,
           const __bf16* __restrict__ W2t,
           const __bf16* __restrict__ W3t,
           const float* __restrict__ b1,
           const float* __restrict__ b2,
           const float* __restrict__ b3,
           float* __restrict__ out) {
    extern __shared__ __align__(32) char smem[];
    __bf16* sIn = (__bf16*)smem;      // [64][256]
    __bf16* sH1 = sIn + 64 * II;      // [64][512]
    __bf16* sH2 = sH1 + 64 * HH;      // [64][512]

    const int tid  = threadIdx.x;
    const int wave = tid >> 5;
    const int lane = tid & 31;
    const int nl   = lane & 15;
    const int mhi  = (lane >> 4) << 3;
    const long tok0 = (long)blockIdx.x * 64;

    // Stage 64x256 bf16 input tile (32 KB) with 16B vectors.
    {
        const uint4* src = (const uint4*)(combined + tok0 * II);
        uint4* dst = (uint4*)sIn;
        const int nvec = (64 * II * 2) / 16;
        for (int i = tid; i < nvec; i += blockDim.x) dst[i] = src[i];
    }
    __syncthreads();

    // ---- Layer 1: [64,256]x[256,512], wave owns N-tiles wave*4..+3 ----
    for (int j = 0; j < 4; ++j) {
        const int n0 = (wave * 4 + j) * 16;
        const float bias = b1[n0 + nl];
        __builtin_prefetch(W1t + (long)(n0 + 16) * II, 0, 1);
        v8f acc[4];
        #pragma unroll
        for (int mt = 0; mt < 4; ++mt)
            acc[mt] = (v8f){0.f, 0.f, 0.f, 0.f, 0.f, 0.f, 0.f, 0.f};
        #pragma unroll
        for (int kk = 0; kk < II; kk += 32) {
            const v16bf b = load_frag(W1t, II, n0, kk);   // loaded once
            #pragma unroll
            for (int mt = 0; mt < 4; ++mt) {              // 4 indep chains
                v16bf a = load_frag(sIn, II, mt * 16, kk);
                acc[mt] = wmma_bf16(a, b, acc[mt]);
            }
        }
        #pragma unroll
        for (int mt = 0; mt < 4; ++mt) {
            #pragma unroll
            for (int e = 0; e < 8; ++e) {
                const int m = mt * 16 + e + mhi;
                float v = acc[mt][e] + bias;
                v = v > 0.0f ? v : 0.0f;
                sH1[m * HH + n0 + nl] = (__bf16)v;
            }
        }
    }
    __syncthreads();

    // ---- Layer 2: [64,512]x[512,512] ----
    for (int j = 0; j < 4; ++j) {
        const int n0 = (wave * 4 + j) * 16;
        const float bias = b2[n0 + nl];
        __builtin_prefetch(W2t + (long)(n0 + 16) * HH, 0, 1);
        v8f acc[4];
        #pragma unroll
        for (int mt = 0; mt < 4; ++mt)
            acc[mt] = (v8f){0.f, 0.f, 0.f, 0.f, 0.f, 0.f, 0.f, 0.f};
        #pragma unroll
        for (int kk = 0; kk < HH; kk += 32) {
            const v16bf b = load_frag(W2t, HH, n0, kk);
            #pragma unroll
            for (int mt = 0; mt < 4; ++mt) {
                v16bf a = load_frag(sH1, HH, mt * 16, kk);
                acc[mt] = wmma_bf16(a, b, acc[mt]);
            }
        }
        #pragma unroll
        for (int mt = 0; mt < 4; ++mt) {
            #pragma unroll
            for (int e = 0; e < 8; ++e) {
                const int m = mt * 16 + e + mhi;
                float v = acc[mt][e] + bias;
                v = v > 0.0f ? v : 0.0f;
                sH2[m * HH + n0 + nl] = (__bf16)v;
            }
        }
    }
    __syncthreads();

    // ---- Layer 3: [64,512]x[512,256], wave owns N-tiles wave*2..+1 ----
    for (int j = 0; j < 2; ++j) {
        const int n0 = (wave * 2 + j) * 16;
        const float bias = b3[n0 + nl];
        __builtin_prefetch(W3t + (long)(n0 + 16) * HH, 0, 1);
        v8f acc[4];
        #pragma unroll
        for (int mt = 0; mt < 4; ++mt)
            acc[mt] = (v8f){0.f, 0.f, 0.f, 0.f, 0.f, 0.f, 0.f, 0.f};
        #pragma unroll
        for (int kk = 0; kk < HH; kk += 32) {
            const v16bf b = load_frag(W3t, HH, n0, kk);
            #pragma unroll
            for (int mt = 0; mt < 4; ++mt) {
                v16bf a = load_frag(sH2, HH, mt * 16, kk);
                acc[mt] = wmma_bf16(a, b, acc[mt]);
            }
        }
        #pragma unroll
        for (int mt = 0; mt < 4; ++mt) {
            #pragma unroll
            for (int e = 0; e < 8; ++e) {
                const int m = mt * 16 + e + mhi;
                out[(tok0 + m) * (long)OO + n0 + nl] = acc[mt][e] + bias;
            }
        }
    }
}

// ---------------------------------------------------------------------------
// Host launcher
// ---------------------------------------------------------------------------
extern "C" void kernel_launch(void* const* d_in, const int* in_sizes, int n_in,
                              void* d_out, int out_size, void* d_ws, size_t ws_size,
                              hipStream_t stream) {
    const float* x  = (const float*)d_in[0];
    const float* Wg = (const float*)d_in[1];
    const float* bg = (const float*)d_in[2];
    const float* W1 = (const float*)d_in[3];
    const float* b1 = (const float*)d_in[4];
    const float* W2 = (const float*)d_in[5];
    const float* b2 = (const float*)d_in[6];
    const float* W3 = (const float*)d_in[7];
    const float* b3 = (const float*)d_in[8];
    float* out = (float*)d_out;

    // Workspace layout (bf16): Wgt | W1t | W2t | W3t | combined
    char* ws = (char*)d_ws;
    __bf16* Wgt = (__bf16*)ws;                 ws += (size_t)II * II * 2;
    __bf16* W1t = (__bf16*)ws;                 ws += (size_t)HH * II * 2;
    __bf16* W2t = (__bf16*)ws;                 ws += (size_t)HH * HH * 2;
    __bf16* W3t = (__bf16*)ws;                 ws += (size_t)OO * HH * 2;
    __bf16* combined = (__bf16*)ws;            // [B*S][I] bf16, 32 MB

    convert_weights<<<512, 256, 0, stream>>>(Wg, W1, W2, W3, Wgt, W1t, W2t, W3t);

    const size_t scan_lds = (size_t)II * II * 2       // Wg bf16
                          + (size_t)2 * 16 * II * 2   // xt bf16 double buffer
                          + (size_t)16 * II * 2       // buf bf16 mirror
                          + (size_t)16 * II * 4;      // buf f32
    scan_kernel<<<BB / 16, 512, scan_lds, stream>>>(x, bg, Wgt, combined);

    const size_t mlp_lds = (size_t)64 * II * 2        // input tile
                         + (size_t)64 * HH * 2        // h1
                         + (size_t)64 * HH * 2;       // h2
    mlp_kernel<<<(BB * SS) / 64, 256, mlp_lds, stream>>>(
        combined, W1t, W2t, W3t, b1, b2, b3, out);
}
